// SSIMLoss_40896678592689
// MI455X (gfx1250) — compile-verified, matrix operand order checked by reading, not run on hardware
//
#include <hip/hip_runtime.h>
#include <hip/hip_bf16.h>

// ---------------------------------------------------------------------------
// SSIM loss on MI455X (gfx1250, wave32).
// Separable 11-tap Gaussian convs cast as 16x16x32 f16 WMMAs (f32 accum).
// One wave per 16x16 output tile; 5 quantities (x, y, x^2, y^2, xy).
// ---------------------------------------------------------------------------

typedef __attribute__((ext_vector_type(16))) _Float16 v16h;
typedef __attribute__((ext_vector_type(8)))  float    v8f;

// Normalized 1-D Gaussian, sigma=1.5, window 11 (matches reference).
__constant__ float G11[11] = {
    0.00102838f, 0.00759880f, 0.03599978f, 0.10936082f, 0.21300053f,
    0.26601172f,
    0.21300053f, 0.10936082f, 0.03599978f, 0.00759880f, 0.00102838f};

#define C1_SSIM 0.0001f   // 0.01^2
#define C2_SSIM 0.0009f   // 0.03^2

#define IMG_W 512
#define IMG_H 512
#define PSTR  34          // LDS stride (f16 elems) = 17 dwords -> conflict-free

// K index held by VGPR pair v (0..7) for lane-half `half` in the 16-bit
// 16x32 A layout (ISA 7.12.2); B 32x16 assumed dual (lane%16 = column N,
// same K mapping).
__device__ __forceinline__ int kbase(int v, int half) {
    return (v < 4) ? (half * 8 + 2 * v) : (16 + half * 8 + 2 * (v - 4));
}

__global__ __launch_bounds__(32) void ssim_tile_kernel(
    const float* __restrict__ img1, const float* __restrict__ img2,
    double* __restrict__ acc)
{
    __shared__ _Float16 sp1[32 * PSTR];   // img1 patch, rows 0..31 (>=26 zero)
    __shared__ _Float16 sp2[32 * PSTR];   // img2 patch
    __shared__ _Float16 shb[16 * PSTR];   // H transpose buffer: [n]*PSTR + k

    const int lane = threadIdx.x;         // 0..31 (one wave)
    const int lm   = lane & 15;
    const int half = lane >> 4;

    const int tile = blockIdx.x;          // 0..1023 within plane
    const int R0   = (tile >> 5) * 16;    // output row base
    const int C0   = (tile & 31) * 16;    // output col base
    const size_t pb = (size_t)blockIdx.y * (IMG_W * IMG_H);

    // --- zero patches (covers conv zero-padding and the K>=26 pad band) ----
    {
        uint32_t* z1 = (uint32_t*)sp1;
        uint32_t* z2 = (uint32_t*)sp2;
        #pragma unroll
        for (int i = 0; i < 17; ++i) {
            z1[i * 32 + lane] = 0u;
            z2[i * 32 + lane] = 0u;
        }
    }
    __syncthreads();

    // --- coalesced patch load: rows R0-5..R0+20, cols C0-5..C0+20 ----------
    for (int i = 0; i < 26; ++i) {
        int gr = R0 - 5 + i;
        int gc = C0 - 5 + lane;
        if (lane < 26 && gr >= 0 && gr < IMG_H && gc >= 0 && gc < IMG_W) {
            size_t o = pb + (size_t)gr * IMG_W + gc;
            sp1[i * PSTR + lane] = (_Float16)img1[o];
            sp2[i * PSTR + lane] = (_Float16)img2[o];
        }
    }
    __syncthreads();

    // --- banded Gaussian weight fragment ----------------------------------
    // As B (horizontal pass): Wh[k][n] = g[k-n], n = lane%16.
    // As A (vertical  pass): Wv[m][k] = g[k-m], m = lane%16.
    // Identical per-lane values under the dual layouts -> one fragment.
    v16h wgt;
    #pragma unroll
    for (int e = 0; e < 16; ++e) {
        int k = kbase(e >> 1, half) + (e & 1);
        int d = k - lm;
        wgt[e] = (_Float16)((d >= 0 && d <= 10) ? G11[d] : 0.0f);
    }

    // --- patch A-fragments (16x32 f16): rows lm / lm+16 -------------------
    union F16x16 { v16h h; uint32_t u[8]; };
    F16x16 a1lo, a1hi, a2lo, a2hi;
    #pragma unroll
    for (int v = 0; v < 8; ++v) {
        int k = kbase(v, half);
        a1lo.u[v] = *(const uint32_t*)&sp1[lm * PSTR + k];
        a1hi.u[v] = *(const uint32_t*)&sp1[(lm + 16) * PSTR + k];
        a2lo.u[v] = *(const uint32_t*)&sp2[lm * PSTR + k];
        a2hi.u[v] = *(const uint32_t*)&sp2[(lm + 16) * PSTR + k];
    }

    // Squares / cross products elementwise on fragments (v_pk_mul_f16 path).
    v16h Alo[5] = { a1lo.h, a2lo.h, a1lo.h * a1lo.h, a2lo.h * a2lo.h,
                    a1lo.h * a2lo.h };
    v16h Ahi[5] = { a1hi.h, a2hi.h, a1hi.h * a1hi.h, a2hi.h * a2hi.h,
                    a1hi.h * a2hi.h };

    v8f outs[5];
    #pragma unroll
    for (int q = 0; q < 5; ++q) {
        v8f zero = {};
        // Horizontal pass: H(26x16) = Patch(26x32) x Wh(32x16), two row blocks.
        v8f h0 = __builtin_amdgcn_wmma_f32_16x16x32_f16(
            false, Alo[q], false, wgt, (short)0, zero, false, false);
        v8f h1 = __builtin_amdgcn_wmma_f32_16x16x32_f16(
            false, Ahi[q], false, wgt, (short)0, zero, false, false);

        __syncthreads();  // WAR on shb vs previous iteration's B-loads
        // D layout -> LDS, column-major [n][k] so K-pairs are contiguous.
        #pragma unroll
        for (int r = 0; r < 8; ++r) {
            int m = r + 8 * half;
            shb[lm * PSTR + m]      = (_Float16)h0[r];
            shb[lm * PSTR + m + 16] = (_Float16)h1[r];
        }
        __syncthreads();

        // Reload H as a 32x16 B-fragment.
        F16x16 hb;
        #pragma unroll
        for (int v = 0; v < 8; ++v) {
            int k = kbase(v, half);
            hb.u[v] = *(const uint32_t*)&shb[lm * PSTR + k];
        }
        // Vertical pass: Out(16x16) = Wv(16x32) x H(32x16).
        outs[q] = __builtin_amdgcn_wmma_f32_16x16x32_f16(
            false, wgt, false, hb.h, (short)0, zero, false, false);
    }

    // --- per-pixel SSIM combine (8 pixels per lane, D layout) --------------
    float lsum = 0.0f;
    #pragma unroll
    for (int r = 0; r < 8; ++r) {
        float mu1 = outs[0][r], mu2 = outs[1][r];
        float e11 = outs[2][r], e22 = outs[3][r], e12 = outs[4][r];
        float mu1sq = mu1 * mu1, mu2sq = mu2 * mu2, mu12 = mu1 * mu2;
        float s1  = fmaxf(e11 - mu1sq, 0.0f);
        float s2  = fmaxf(e22 - mu2sq, 0.0f);
        float s12 = e12 - mu12;
        float num = (2.0f * mu12 + C1_SSIM) * (2.0f * s12 + C2_SSIM);
        float den = (mu1sq + mu2sq + C1_SSIM) * (s1 + s2 + C2_SSIM);
        float v   = num / den;
        lsum += fminf(fmaxf(v, 0.0f), 1.0f);
    }

    // Wave32 reduction + one f64 atomic per tile.
    #pragma unroll
    for (int o = 16; o > 0; o >>= 1)
        lsum += __shfl_xor(lsum, o, 32);
    if (lane == 0)
        atomicAdd(acc, (double)lsum);
}

__global__ void ssim_init_kernel(double* acc) { *acc = 0.0; }

__global__ void ssim_final_kernel(const double* acc, float* out, double invN) {
    out[0] = 1.0f - (float)(*acc * invN);
}

extern "C" void kernel_launch(void* const* d_in, const int* in_sizes, int n_in,
                              void* d_out, int out_size, void* d_ws, size_t ws_size,
                              hipStream_t stream) {
    const float* img1 = (const float*)d_in[0];
    const float* img2 = (const float*)d_in[1];
    double* acc = (double*)d_ws;

    const long long N = (long long)in_sizes[0];         // B*C*H*W
    const int planes  = (int)(N / (IMG_W * IMG_H));     // B*C = 96
    const int tiles   = (IMG_W / 16) * (IMG_H / 16);    // 1024 per plane

    ssim_init_kernel<<<1, 1, 0, stream>>>(acc);
    ssim_tile_kernel<<<dim3(tiles, planes), 32, 0, stream>>>(img1, img2, acc);
    ssim_final_kernel<<<1, 1, 0, stream>>>(acc, (float*)d_out, 1.0 / (double)N);
}
